// MultiHeadAttention_17523466568217
// MI455X (gfx1250) — matrix operand order repaired and checked by reading, not compiled, not run on hardware
//
#include <hip/hip_runtime.h>
#include <hip/hip_bf16.h>
#include <stdint.h>

// ---------------------------------------------------------------------------
// MHA forward for MI455X (gfx1250, wave32, WMMA 16x16x32 bf16 / f32 acc)
//   B=2, T=2048, D=2048, H=16, dh=128
// Pipeline: cast/transposes -> QKV GEMM(+split,scale,V-transpose) ->
//           flash attention -> output projection GEMM
// Shared operand tiles are staged per-block into LDS with the CDNA5 async
// copy path (global_load_async_to_lds_b128 / ASYNCcnt), double-buffered so
// the next tile's DMA overlaps the current tile's WMMAs.  Fragment loads are
// issued in bulk ahead of the WMMA chain so LDS latency pipelines.
// ---------------------------------------------------------------------------

typedef __attribute__((ext_vector_type(16))) __bf16 v16bf;
typedef __attribute__((ext_vector_type(8)))  __bf16 v8bf;
typedef __attribute__((ext_vector_type(8)))  float  v8f;

#define D_MODEL 2048
#define NHEAD   16
#define D_HEAD  128
#define BATCH   2
#define SEQ     2048
#define M_TOT   (BATCH * SEQ)      // 4096 rows
#define N_QKV   (3 * D_MODEL)      // 6144 cols

static __device__ __forceinline__ __bf16 f2bf(float f) {
    uint32_t u = __builtin_bit_cast(uint32_t, f);
    u += 0x7fffu + ((u >> 16) & 1u);                 // round-to-nearest-even
    return __builtin_bit_cast(__bf16, (uint16_t)(u >> 16));
}

// Per the ISA aperture rules, addr[31:0] of a generic LDS pointer is the
// wave-relative LDS byte offset, which is exactly what the async VDST wants.
static __device__ __forceinline__ uint32_t lds_off(const void* p) {
    return (uint32_t)(uintptr_t)p;
}

// Async DMA: 16 bytes global -> LDS, tracked by ASYNCcnt (no VGPR transit).
static __device__ __forceinline__ void async_copy16(uint32_t lds, const void* g) {
    asm volatile("global_load_async_to_lds_b128 %0, %1, off"
                 :: "v"(lds), "v"((uint64_t)(uintptr_t)g)
                 : "memory");
}
static __device__ __forceinline__ void wait_async0() {
    asm volatile("s_wait_asynccnt 0" ::: "memory");
}

// A-operand fragment (16x32 bf16). rowbase = &src[row][0] (row-major, row = lane&15).
// ISA layout: element e -> K = k0 + 16*(e>>3) + 8*half + (e&7)
static __device__ __forceinline__ v16bf load_fragA(const __bf16* rowbase, int k0, int half) {
    const int o = k0 + 8 * half;
    v8bf lo = *(const v8bf*)(rowbase + o);
    v8bf hi = *(const v8bf*)(rowbase + o + 16);
    return __builtin_shufflevector(lo, hi, 0,1,2,3,4,5,6,7,8,9,10,11,12,13,14,15);
}

// B-operand fragment (32x16 bf16). Source is N-major: rowbase = &Bt[n][0], n = lane&15 col.
// ISA layout: element e -> K = k0 + 16*(e>>3) + 8*half + (e&7) (same as A).
static __device__ __forceinline__ v16bf load_fragB(const __bf16* rowbase, int k0, int half) {
    const int o = k0 + 8 * half;
    v8bf lo = *(const v8bf*)(rowbase + o);
    v8bf hi = *(const v8bf*)(rowbase + o + 16);
    return __builtin_shufflevector(lo, hi, 0,1,2,3,4,5,6,7,8,9,10,11,12,13,14,15);
}

static __device__ __forceinline__ v8f wmma_bf16(v16bf a, v16bf b, v8f c) {
    return __builtin_amdgcn_wmma_f32_16x16x32_bf16(false, a, false, b, (short)0, c, false, false);
}

// --------------------------- elementwise prep ------------------------------

__global__ void k_cast_bf16(const float* __restrict__ in, __bf16* __restrict__ out, int n) {
    int i = blockIdx.x * blockDim.x + threadIdx.x;
    if (i < n) out[i] = f2bf(in[i]);
}

// W[K][N] (f32) -> Wt[N][K] (bf16)
__global__ void k_transpose_bf16(const float* __restrict__ W, __bf16* __restrict__ Wt,
                                 int K, int N) {
    int i = blockIdx.x * blockDim.x + threadIdx.x;
    if (i >= K * N) return;
    int n = i % N, k = i / N;
    Wt[(size_t)n * K + k] = f2bf(W[(size_t)k * N + n]);
}

// ------------------------------- GEMM --------------------------------------
// C[M,N] = A[M,K] * Bt[N,K]^T + bias.  8 waves/block, wave tile = 16x64.
// The 64x32 B tile (shared by all 8 waves) is async-staged into LDS,
// double-buffered.  A fragments (private rows) load straight from global.
// mode 0: write f32 to outF.
// mode 1: QKV epilogue: split cols into q (x0.25, bf16 [bh][t][d]),
//         k (bf16 [bh][t][d]), v transposed (bf16 [bh][d][t]).
__global__ void __launch_bounds__(256)
k_gemm_bf16(const __bf16* __restrict__ A, const __bf16* __restrict__ Bt,
            const float* __restrict__ bias, int M, int N, int K, int mode,
            float* __restrict__ outF, __bf16* __restrict__ outQ,
            __bf16* __restrict__ outK, __bf16* __restrict__ outVT)
{
    __shared__ alignas(16) __bf16 bTile[2][64][32];   // 8 KB, double-buffered

    const int tid  = threadIdx.x;
    const int lane = tid & 31;
    const int wv   = tid >> 5;
    const int half = lane >> 4;
    const int ln   = lane & 15;

    const int r0 = blockIdx.y * 128 + wv * 16;
    const int c0 = blockIdx.x * 64;

    const __bf16* arow = A + (size_t)(r0 + ln) * K;

    // one 16B chunk per thread per k-step: col = tid>>2, 16B part = tid&3
    const int stCol  = tid >> 2;
    const int stPart = tid & 3;
    const __bf16* stSrcRow = Bt + (size_t)(c0 + stCol) * K + stPart * 8;
    auto stageB = [&](int buf, int k0) {
        async_copy16(lds_off(&bTile[buf][stCol][stPart * 8]), stSrcRow + k0);
    };

    v8f acc[4];
#pragma unroll
    for (int j = 0; j < 4; ++j) acc[j] = (v8f){0,0,0,0,0,0,0,0};

    const int nIter = K / 32;
    stageB(0, 0);
    for (int it = 0; it < nIter; ++it) {
        const int buf = it & 1;
        wait_async0();          // our chunk of bTile[buf] landed
        __syncthreads();        // everyone's chunks visible; prev buffer consumed
        if (it + 1 < nIter) stageB(buf ^ 1, (it + 1) * 32);

        // issue every fragment load for this tile before the WMMA chain
        v16bf a = load_fragA(arow, it * 32, half);
        v16bf b[4];
#pragma unroll
        for (int j = 0; j < 4; ++j)
            b[j] = load_fragB(&bTile[buf][j * 16 + ln][0], 0, half);
#pragma unroll
        for (int j = 0; j < 4; ++j)
            acc[j] = wmma_bf16(a, b[j], acc[j]);
        __syncthreads();        // done reading bTile[buf] before it is restaged
    }

#pragma unroll
    for (int j = 0; j < 4; ++j) {
        const int n  = c0 + j * 16 + ln;
        const float bv = bias ? bias[n] : 0.0f;
#pragma unroll
        for (int v = 0; v < 8; ++v) {
            const int r = r0 + v + 8 * half;       // C layout: row = v + 8*half
            float val = acc[j][v] + bv;
            if (mode == 0) {
                outF[(size_t)r * N + n] = val;
            } else {
                const int b_  = r >> 11;           // / SEQ
                const int t   = r & (SEQ - 1);
                const int wch = n >> 11;           // 0:q 1:k 2:v
                const int h   = (n & (D_MODEL - 1)) >> 7;
                const int d   = n & (D_HEAD - 1);
                const size_t bh = (size_t)(b_ * NHEAD + h);
                if (wch == 0)
                    outQ[(bh * SEQ + t) * D_HEAD + d] = f2bf(val * 0.25f); // fold score scale
                else if (wch == 1)
                    outK[(bh * SEQ + t) * D_HEAD + d] = f2bf(val);
                else
                    outVT[(bh * D_HEAD + d) * SEQ + t] = f2bf(val);        // V transposed
            }
        }
    }
}

// --------------------------- flash attention -------------------------------
// One wave = 16 query rows x full dh=128. grid = (SEQ/128, BATCH*NHEAD), 256 thr.
// K tile (32 x 128) and V tile (128 x 32, from transposed V) are shared by all
// 8 waves -> async-staged into LDS, double-buffered.
__global__ void __launch_bounds__(256)
k_flash_attn(const __bf16* __restrict__ Q, const __bf16* __restrict__ Kh,
             const __bf16* __restrict__ VT, __bf16* __restrict__ X)
{
    __shared__ alignas(16) __bf16 kTile[2][32][128];  // 16 KB
    __shared__ alignas(16) __bf16 vTile[2][128][32];  // 16 KB
    __shared__ alignas(16) __bf16 pTile[8][16][32];   //  8 KB P re-layout

    const int tid  = threadIdx.x;
    const int lane = tid & 31;
    const int wv   = tid >> 5;
    const int half = lane >> 4;
    const int ln   = lane & 15;

    const int bh = blockIdx.y;                 // b*NHEAD + h
    const int t0 = blockIdx.x * 128 + wv * 16;

    const __bf16* qbase = Q  + (size_t)bh * SEQ * D_HEAD;
    const __bf16* kbase = Kh + (size_t)bh * SEQ * D_HEAD;
    const __bf16* vbase = VT + (size_t)bh * D_HEAD * SEQ;

    auto stageKV = [&](int buf, int s0) {
        // K rows s0..s0+31 are one contiguous 8 KB block: 512 x 16B chunks
#pragma unroll
        for (int c = tid; c < 512; c += 256)
            async_copy16(lds_off(&kTile[buf][0][0] + c * 8),
                         kbase + (size_t)s0 * D_HEAD + c * 8);
        // V^T rows d=0..127, 64B (32 bf16) each at column block s0
#pragma unroll
        for (int c = tid; c < 512; c += 256) {
            const int d = c >> 2, part = c & 3;
            async_copy16(lds_off(&vTile[buf][d][part * 8]),
                         vbase + (size_t)d * SEQ + s0 + part * 8);
        }
    };

    // Q fragments for this wave's 16 rows (scale pre-folded at QKV epilogue)
    const __bf16* qrow = qbase + (size_t)(t0 + ln) * D_HEAD;
    v16bf qa[4];
#pragma unroll
    for (int i = 0; i < 4; ++i) qa[i] = load_fragA(qrow, 32 * i, half);

    v8f O[8];
#pragma unroll
    for (int j = 0; j < 8; ++j) O[j] = (v8f){0,0,0,0,0,0,0,0};
    float mrow[8], lrow[8];
#pragma unroll
    for (int v = 0; v < 8; ++v) { mrow[v] = -1e30f; lrow[v] = 0.0f; }

    const int nIter = SEQ / 32;
    stageKV(0, 0);
    for (int it = 0; it < nIter; ++it) {
        const int buf = it & 1;
        wait_async0();
        __syncthreads();
        if (it + 1 < nIter) stageKV(buf ^ 1, (it + 1) * 32);

        // S = Q * K^T for 32 score columns (two 16x16 WMMA tiles, K-loop over dh).
        // Issue all 8 K-fragment loads first so the ds latency pipelines.
        v16bf kb1[4], kb2[4];
#pragma unroll
        for (int i = 0; i < 4; ++i) {
            kb1[i] = load_fragB(&kTile[buf][ln][0],      32 * i, half);
            kb2[i] = load_fragB(&kTile[buf][16 + ln][0], 32 * i, half);
        }
        v8f S1 = (v8f){0,0,0,0,0,0,0,0};
        v8f S2 = (v8f){0,0,0,0,0,0,0,0};
#pragma unroll
        for (int i = 0; i < 4; ++i) {
            S1 = wmma_bf16(qa[i], kb1[i], S1);
            S2 = wmma_bf16(qa[i], kb2[i], S2);
        }

        // online softmax: C-layout row (v + 8*half) lives across 16 lanes
        float P1[8], P2[8];
#pragma unroll
        for (int v = 0; v < 8; ++v) {
            float mx = fmaxf(S1[v], S2[v]);
#pragma unroll
            for (int msk = 1; msk < 16; msk <<= 1)
                mx = fmaxf(mx, __shfl_xor(mx, msk, 32));
            const float mn    = fmaxf(mrow[v], mx);
            const float alpha = __expf(mrow[v] - mn);
            const float p1 = __expf(S1[v] - mn);
            const float p2 = __expf(S2[v] - mn);
            float ps = p1 + p2;
#pragma unroll
            for (int msk = 1; msk < 16; msk <<= 1)
                ps += __shfl_xor(ps, msk, 32);
            lrow[v] = lrow[v] * alpha + ps;
            mrow[v] = mn;
            P1[v] = p1;
            P2[v] = p2;
#pragma unroll
            for (int j = 0; j < 8; ++j) O[j][v] *= alpha;
        }

        // C-layout -> A-fragment re-layout through this wave's private LDS tile
#pragma unroll
        for (int v = 0; v < 8; ++v) {
            pTile[wv][v + 8 * half][ln]      = f2bf(P1[v]);
            pTile[wv][v + 8 * half][16 + ln] = f2bf(P2[v]);
        }
        // same-wave DS ops are architecturally in-order; the wait + memory
        // clobber stops the compiler from hoisting the cross-lane reads.
        asm volatile("s_wait_dscnt 0" ::: "memory");
        v16bf pa = load_fragA(&pTile[wv][ln][0], 0, half);

        // O += P * V  (V tile is [d][s] -> B-fragment rows contiguous, k0 = 0).
        // Bulk-issue all 8 V-fragment loads, then the WMMA chain.
        v16bf vb[8];
#pragma unroll
        for (int j = 0; j < 8; ++j)
            vb[j] = load_fragB(&vTile[buf][j * 16 + ln][0], 0, half);
#pragma unroll
        for (int j = 0; j < 8; ++j)
            O[j] = wmma_bf16(pa, vb[j], O[j]);
        __syncthreads();        // done reading kTile/vTile[buf] before restage
    }

    // normalize and scatter to x[b][t][h*dh + d] (bf16)
    const int b_ = bh >> 4, h = bh & 15;
#pragma unroll
    for (int j = 0; j < 8; ++j) {
#pragma unroll
        for (int v = 0; v < 8; ++v) {
            const int t = t0 + v + 8 * half;
            const float o = O[j][v] / lrow[v];
            X[((size_t)(b_ * SEQ + t)) * D_MODEL + h * D_HEAD + j * 16 + ln] = f2bf(o);
        }
    }
}

// ------------------------------- launcher ----------------------------------

extern "C" void kernel_launch(void* const* d_in, const int* in_sizes, int n_in,
                              void* d_out, int out_size, void* d_ws, size_t ws_size,
                              hipStream_t stream) {
    (void)in_sizes; (void)n_in; (void)out_size; (void)ws_size;

    const float* query = (const float*)d_in[0];
    // d_in[1] (key) and d_in[2] (value) are unused by the reference module.
    const float* W_qkv = (const float*)d_in[3];
    const float* b_qkv = (const float*)d_in[4];
    const float* W_out = (const float*)d_in[5];
    const float* b_out = (const float*)d_in[6];
    float* out = (float*)d_out;

    char*  ws  = (char*)d_ws;
    size_t off = 0;
    auto alloc = [&](size_t bytes) -> void* {
        void* p = ws + off;
        off += (bytes + 255) & ~(size_t)255;
        return p;
    };
    __bf16* qbf   = (__bf16*)alloc((size_t)M_TOT * D_MODEL * 2);            // 16.8 MB
    __bf16* wqkvT = (__bf16*)alloc((size_t)N_QKV * D_MODEL * 2);            // 25.2 MB
    __bf16* woutT = (__bf16*)alloc((size_t)D_MODEL * D_MODEL * 2);          //  8.4 MB
    __bf16* qh    = (__bf16*)alloc((size_t)BATCH * NHEAD * SEQ * D_HEAD * 2);
    __bf16* kh    = (__bf16*)alloc((size_t)BATCH * NHEAD * SEQ * D_HEAD * 2);
    __bf16* vT    = (__bf16*)alloc((size_t)BATCH * NHEAD * D_HEAD * SEQ * 2);
    __bf16* xbf   = (__bf16*)alloc((size_t)M_TOT * D_MODEL * 2);            // 16.8 MB

    {   // cast activations
        int n = M_TOT * D_MODEL;
        k_cast_bf16<<<(n + 255) / 256, 256, 0, stream>>>(query, qbf, n);
    }
    {   // transpose weights to [N][K] bf16
        int n = D_MODEL * N_QKV;
        k_transpose_bf16<<<(n + 255) / 256, 256, 0, stream>>>(W_qkv, wqkvT, D_MODEL, N_QKV);
        n = D_MODEL * D_MODEL;
        k_transpose_bf16<<<(n + 255) / 256, 256, 0, stream>>>(W_out, woutT, D_MODEL, D_MODEL);
    }
    {   // QKV projection + head split + q-scale + V-transpose
        dim3 g(N_QKV / 64, M_TOT / 128);
        k_gemm_bf16<<<g, 256, 0, stream>>>(qbf, wqkvT, b_qkv, M_TOT, N_QKV, D_MODEL, 1,
                                           nullptr, qh, kh, vT);
    }
    {   // flash attention
        dim3 g(SEQ / 128, BATCH * NHEAD);
        k_flash_attn<<<g, 256, 0, stream>>>(qh, kh, vT, xbf);
    }
    {   // output projection -> f32 result
        dim3 g(D_MODEL / 64, M_TOT / 128);
        k_gemm_bf16<<<g, 256, 0, stream>>>(xbf, woutT, b_out, M_TOT, D_MODEL, D_MODEL, 0,
                                           out, nullptr, nullptr, nullptr);
    }
}
